// SelfAttention_69578470195644
// MI455X (gfx1250) — matrix operand order repaired and checked by reading, not compiled, not run on hardware
//
#include <hip/hip_runtime.h>
#include <hip/hip_bf16.h>
#include <math.h>

// ---------------------------------------------------------------------------
// Types for gfx1250 WMMA (wave32)
// ---------------------------------------------------------------------------
typedef __bf16 bf16;
typedef __attribute__((ext_vector_type(16))) __bf16 v16bf;
typedef __attribute__((ext_vector_type(8)))  __bf16 v8bf;
typedef __attribute__((ext_vector_type(8)))  float   v8f;

#define S_LEN 2048
#define DIM_  2048
#define NH    32
#define NKV   8
#define HD_   64
#define SCALE 0.125f   // HD^-0.5, folded into Q during RoPE

__device__ __forceinline__ v8f vzero8f() {
    v8f z = {0.f,0.f,0.f,0.f,0.f,0.f,0.f,0.f};
    return z;
}

// A fragment, 16x32 bf16, source row-major with leading dim `ld`.
// Layout used by this code (same k order as B): lane L holds row M=L&15;
// half h=L>>4: elements 0..15 -> K=16h+0..15 (contiguous, matches B).
__device__ __forceinline__ v16bf load_a_frag(const bf16* p, int ld, int lane) {
    const int m = lane & 15, h = lane >> 4;
    const bf16* base = p + (size_t)m * ld + 16 * h;
    v8bf lo = *(const v8bf*)(base);        // K = 16h .. 16h+7
    v8bf hi = *(const v8bf*)(base + 8);    // K = 16h+8 .. 16h+15
    v16bf r;
#pragma unroll
    for (int i = 0; i < 8; ++i) { r[i] = lo[i]; r[i + 8] = hi[i]; }
    return r;
}

// B fragment for C = A * Bsrc^T, Bsrc row-major [N,K] with leading dim ld.
// WMMA B[k,n] = Bsrc[n,k]. Lane L: column n=L&15 fixed, K = 16h+0..15
// -> one contiguous 32B load per lane.
__device__ __forceinline__ v16bf load_bt_frag(const bf16* p, int ld, int lane) {
    const int n = lane & 15, h = lane >> 4;
    return *(const v16bf*)(p + (size_t)n * ld + 16 * h);
}

__device__ __forceinline__ v8f wmma_bf16(v16bf a, v16bf b, v8f c) {
    return __builtin_amdgcn_wmma_f32_16x16x32_bf16(
        /*neg_a=*/false, a, /*neg_b=*/false, b,
        /*c_mod=*/(short)0, c, /*reuse_a=*/false, /*reuse_b=*/false);
}

// ---------------------------------------------------------------------------
// fp32 -> bf16 conversion
// ---------------------------------------------------------------------------
__global__ void f32_to_bf16_kernel(const float* __restrict__ src,
                                   bf16* __restrict__ dst, int n) {
    int i = blockIdx.x * blockDim.x + threadIdx.x;
    if (i < n) dst[i] = (bf16)src[i];
}

// ---------------------------------------------------------------------------
// Generic GEMM-T: C[M,N] = A[M,K] * B[N,K]^T. A,B bf16 row-major.
// One wave computes a 32x32 tile via 2x2 WMMA accumulators.
// OUT_MODE: 0 = fp32 row-major, 1 = bf16 row-major, 2 = bf16 transposed (C^T, ld=M)
// ---------------------------------------------------------------------------
template <int OUT_MODE>
__global__ void gemm_bt_kernel(const bf16* __restrict__ A, const bf16* __restrict__ B,
                               void* __restrict__ Out, int M, int N, int K) {
    const int lane = threadIdx.x;
    const int n0 = blockIdx.x * 32;
    const int m0 = blockIdx.y * 32;

    v8f c00 = vzero8f(), c01 = vzero8f(), c10 = vzero8f(), c11 = vzero8f();

    const bf16* a0p = A + (size_t)m0 * K;
    const bf16* a1p = A + (size_t)(m0 + 16) * K;
    const bf16* b0p = B + (size_t)n0 * K;
    const bf16* b1p = B + (size_t)(n0 + 16) * K;

    for (int k = 0; k < K; k += 32) {
        __builtin_prefetch(a0p + k + 128, 0, 1);   // global_prefetch_b8
        __builtin_prefetch(b0p + k + 128, 0, 1);
        v16bf a0 = load_a_frag(a0p + k, K, lane);
        v16bf a1 = load_a_frag(a1p + k, K, lane);
        v16bf b0 = load_bt_frag(b0p + k, K, lane);
        v16bf b1 = load_bt_frag(b1p + k, K, lane);
        c00 = wmma_bf16(a0, b0, c00);
        c01 = wmma_bf16(a0, b1, c01);
        c10 = wmma_bf16(a1, b0, c10);
        c11 = wmma_bf16(a1, b1, c11);
    }

    const int col = lane & 15, h = lane >> 4;
#pragma unroll
    for (int r = 0; r < 8; ++r) {
        const int mr0 = m0 + r + 8 * h;   // C layout: VGPR r -> row r + 8*half
        const int mr1 = mr0 + 16;
        if (OUT_MODE == 0) {
            float* O = (float*)Out;
            O[(size_t)mr0 * N + n0 + col]      = c00[r];
            O[(size_t)mr0 * N + n0 + 16 + col] = c01[r];
            O[(size_t)mr1 * N + n0 + col]      = c10[r];
            O[(size_t)mr1 * N + n0 + 16 + col] = c11[r];
        } else if (OUT_MODE == 1) {
            bf16* O = (bf16*)Out;
            O[(size_t)mr0 * N + n0 + col]      = (bf16)c00[r];
            O[(size_t)mr0 * N + n0 + 16 + col] = (bf16)c01[r];
            O[(size_t)mr1 * N + n0 + col]      = (bf16)c10[r];
            O[(size_t)mr1 * N + n0 + 16 + col] = (bf16)c11[r];
        } else {
            bf16* O = (bf16*)Out;             // transposed: O[n, m], ld = M
            O[(size_t)(n0 + col) * M + mr0]      = (bf16)c00[r];
            O[(size_t)(n0 + 16 + col) * M + mr0] = (bf16)c01[r];
            O[(size_t)(n0 + col) * M + mr1]      = (bf16)c10[r];
            O[(size_t)(n0 + 16 + col) * M + mr1] = (bf16)c11[r];
        }
    }
}

// ---------------------------------------------------------------------------
// RoPE (in place, bf16), with optional output scale (1/sqrt(HD) folded into Q).
// freqs: [S, HD] fp32; freq for pair i is freqs[s*HD + i], i < HD/2.
// ---------------------------------------------------------------------------
__global__ void rope_kernel(bf16* __restrict__ xh, int nheads,
                            const float* __restrict__ freqs, float oscale) {
    int idx = blockIdx.x * blockDim.x + threadIdx.x;
    int i  = idx & 31;                       // pair index 0..31
    int hh = (idx >> 5) % nheads;
    int s  = idx / (32 * nheads);
    if (s >= S_LEN) return;
    float f = freqs[s * HD_ + i];
    float c = cosf(f), sn = sinf(f);
    bf16* p = xh + (size_t)s * nheads * HD_ + hh * HD_;
    float xe = (float)p[2 * i];
    float xo = (float)p[2 * i + 1];
    p[2 * i]     = (bf16)((xe * c - xo * sn) * oscale);
    p[2 * i + 1] = (bf16)((xo * c + xe * sn) * oscale);
}

// ---------------------------------------------------------------------------
// Flash attention: one wave per (head, 16 query rows). Causal, GQA 4:1.
// Q: [S, H*HD] bf16 (roped, pre-scaled)   K: [S, KV*HD] bf16 (roped)
// Vt: [KV*HD, S] bf16 (V transposed)      Ctx: [S, H*HD] bf16
// Softmax reductions are issued stage-major so the 8 independent row
// butterflies pipeline (8 ds_bpermute in flight per s_wait_dscnt, not 1).
// ---------------------------------------------------------------------------
__global__ void attn_kernel(const bf16* __restrict__ Q, const bf16* __restrict__ Kb,
                            const bf16* __restrict__ Vt, bf16* __restrict__ Ctx) {
    __shared__ bf16 Plds[16 * 40];           // 16x32 P tile, padded ld=40

    const int lane = threadIdx.x;
    const int m0   = blockIdx.x * 16;        // query block (16 rows)
    const int head = blockIdx.y;
    const int kvh  = head >> 2;              // H/KV = 4
    const int col  = lane & 15, h = lane >> 4;

    v8f acc[4];
    float rmax[8], rsum[8];
#pragma unroll
    for (int id = 0; id < 4; ++id) acc[id] = vzero8f();
#pragma unroll
    for (int r = 0; r < 8; ++r) { rmax[r] = -3.0e38f; rsum[r] = 0.f; }

    // Q fragments are invariant across the key loop: load once.
    const bf16* qbase = Q + (size_t)m0 * DIM_ + head * HD_;
    const v16bf qa0 = load_a_frag(qbase +  0, DIM_, lane);
    const v16bf qa1 = load_a_frag(qbase + 32, DIM_, lane);

    const bf16* kbase  = Kb + kvh * HD_;
    const bf16* vtbase = Vt + (size_t)(kvh * HD_) * S_LEN;

    for (int t0 = 0; t0 <= m0 + 15; t0 += 32) {
        // ---- S = Q * K^T (16x32 tile), K-dim = HD = 64 -> 2 steps ----
        v8f s0 = vzero8f(), s1 = vzero8f();
        {
            const bf16* k0p = kbase + (size_t)t0 * (NKV * HD_);
            const bf16* k1p = kbase + (size_t)(t0 + 16) * (NKV * HD_);
            v16bf b00 = load_bt_frag(k0p +  0, NKV * HD_, lane);
            v16bf b01 = load_bt_frag(k1p +  0, NKV * HD_, lane);
            v16bf b10 = load_bt_frag(k0p + 32, NKV * HD_, lane);
            v16bf b11 = load_bt_frag(k1p + 32, NKV * HD_, lane);
            s0 = wmma_bf16(qa0, b00, s0);
            s1 = wmma_bf16(qa0, b01, s1);
            s0 = wmma_bf16(qa1, b10, s0);
            s1 = wmma_bf16(qa1, b11, s1);
        }

        // ---- causal mask: only the diagonal block needs it ----
        if (t0 + 31 > m0) {                  // uniform (scalar) branch
#pragma unroll
            for (int r = 0; r < 8; ++r) {
                const int m = m0 + r + 8 * h;
                s0[r] = (t0 + col      <= m) ? s0[r] : -1.0e30f;
                s1[r] = (t0 + 16 + col <= m) ? s1[r] : -1.0e30f;
            }
        }

        // ---- online softmax, stage-major butterflies (row = r + 8h) ----
        float mx[8], ps[8], corr[8];
#pragma unroll
        for (int r = 0; r < 8; ++r)
            mx[r] = fmaxf(rmax[r], fmaxf(s0[r], s1[r]));
#pragma unroll
        for (int d = 1; d <= 8; d <<= 1) {   // 8 independent permutes per stage
#pragma unroll
            for (int r = 0; r < 8; ++r)
                mx[r] = fmaxf(mx[r], __shfl_xor(mx[r], d, 32));
        }
#pragma unroll
        for (int r = 0; r < 8; ++r) {
            corr[r] = __expf(rmax[r] - mx[r]);
            const float p0 = __expf(s0[r] - mx[r]);
            const float p1 = __expf(s1[r] - mx[r]);
            const int rl = r + 8 * h;
            Plds[rl * 40 + col]      = (bf16)p0;
            Plds[rl * 40 + 16 + col] = (bf16)p1;
            ps[r] = p0 + p1;
            rmax[r] = mx[r];
        }
#pragma unroll
        for (int d = 1; d <= 8; d <<= 1) {
#pragma unroll
            for (int r = 0; r < 8; ++r)
                ps[r] += __shfl_xor(ps[r], d, 32);
        }
#pragma unroll
        for (int r = 0; r < 8; ++r) {
            rsum[r] = rsum[r] * corr[r] + ps[r];
#pragma unroll
            for (int id = 0; id < 4; ++id)
                acc[id][r] = acc[id][r] * corr[r];
        }
        __syncthreads();   // LDS visibility fence (cross-lane P relayout)

        // ---- acc += P * V  (P: 16x32 from LDS in A-layout, V from Vt rows) ----
        const v16bf pa = load_a_frag(Plds, 40, lane);
#pragma unroll
        for (int id = 0; id < 4; ++id) {
            v16bf bv = load_bt_frag(vtbase + (size_t)(id * 16) * S_LEN + t0,
                                    S_LEN, lane);
            acc[id] = wmma_bf16(pa, bv, acc[id]);
        }
        __syncthreads();
    }

    // ---- normalize and store context ----
#pragma unroll
    for (int r = 0; r < 8; ++r) {
        const float inv = 1.0f / rsum[r];
        const int m = m0 + r + 8 * h;
#pragma unroll
        for (int id = 0; id < 4; ++id)
            Ctx[(size_t)m * DIM_ + head * HD_ + id * 16 + col] =
                (bf16)(acc[id][r] * inv);
    }
}

// ---------------------------------------------------------------------------
// Host-side launcher
// ---------------------------------------------------------------------------
extern "C" void kernel_launch(void* const* d_in, const int* in_sizes, int n_in,
                              void* d_out, int out_size, void* d_ws, size_t ws_size,
                              hipStream_t stream) {
    const float* x     = (const float*)d_in[0];
    // d_in[1] = mask (causal, recomputed analytically)
    const float* wq    = (const float*)d_in[2];
    const float* wk    = (const float*)d_in[3];
    const float* wv    = (const float*)d_in[4];
    const float* wo    = (const float*)d_in[5];
    const float* freqs = (const float*)d_in[6];
    float* out = (float*)d_out;

    // Workspace layout (bf16 buffers), ~48 MB total
    char* ws = (char*)d_ws;
    const size_t SZ_XD = (size_t)S_LEN * DIM_;        // 4M elems
    const size_t SZ_KV = (size_t)(NKV * HD_) * DIM_;  // 1M elems
    bf16* xb  = (bf16*)ws;                 ws += SZ_XD * 2;
    bf16* wqb = (bf16*)ws;                 ws += SZ_XD * 2;
    bf16* wkb = (bf16*)ws;                 ws += SZ_KV * 2;
    bf16* wvb = (bf16*)ws;                 ws += SZ_KV * 2;
    bf16* wob = (bf16*)ws;                 ws += SZ_XD * 2;
    bf16* qb  = (bf16*)ws;                 ws += SZ_XD * 2;   // [S, H*HD]
    bf16* kb  = (bf16*)ws;                 ws += (size_t)S_LEN * NKV * HD_ * 2; // [S, KV*HD]
    bf16* vT  = (bf16*)ws;                 ws += (size_t)NKV * HD_ * S_LEN * 2; // [KV*HD, S]
    bf16* ctx = (bf16*)ws;                 ws += SZ_XD * 2;   // [S, H*HD]

    // 1) fp32 -> bf16 conversions
    auto conv = [&](const float* s, bf16* d, int n) {
        f32_to_bf16_kernel<<<(n + 255) / 256, 256, 0, stream>>>(s, d, n);
    };
    conv(x,  xb,  (int)SZ_XD);
    conv(wq, wqb, (int)SZ_XD);
    conv(wk, wkb, (int)SZ_KV);
    conv(wv, wvb, (int)SZ_KV);
    conv(wo, wob, (int)SZ_XD);

    // 2) Q/K/V projections (all C = X * W^T)
    gemm_bt_kernel<1><<<dim3(DIM_ / 32, S_LEN / 32), 32, 0, stream>>>(
        xb, wqb, qb, S_LEN, DIM_, DIM_);
    gemm_bt_kernel<1><<<dim3((NKV * HD_) / 32, S_LEN / 32), 32, 0, stream>>>(
        xb, wkb, kb, S_LEN, NKV * HD_, DIM_);
    gemm_bt_kernel<2><<<dim3((NKV * HD_) / 32, S_LEN / 32), 32, 0, stream>>>(
        xb, wvb, vT, S_LEN, NKV * HD_, DIM_);   // written transposed

    // 3) RoPE on Q (with 1/sqrt(HD) folded in) and K
    {
        int nq = S_LEN * NH * 32;
        rope_kernel<<<(nq + 255) / 256, 256, 0, stream>>>(qb, NH, freqs, SCALE);
        int nk = S_LEN * NKV * 32;
        rope_kernel<<<(nk + 255) / 256, 256, 0, stream>>>(kb, NKV, freqs, 1.0f);
    }

    // 4) Attention (flash, causal, GQA) — 16 query rows per wave
    attn_kernel<<<dim3(S_LEN / 16, NH), 32, 0, stream>>>(qb, kb, vT, ctx);

    // 5) Output projection -> fp32 d_out
    gemm_bt_kernel<0><<<dim3(DIM_ / 32, S_LEN / 32), 32, 0, stream>>>(
        ctx, wob, out, S_LEN, DIM_, DIM_);

    (void)in_sizes; (void)n_in; (void)out_size; (void)ws_size;
}
